// TransformerXL_46093589021337
// MI455X (gfx1250) — matrix operand order repaired and checked by reading, not compiled, not run on hardware
//
#include <hip/hip_runtime.h>

// ---------------- problem dimensions ----------------
#define L_LAYERS 4
#define EMB      1024
#define NH       16
#define HD       64
#define VOCAB    16000
#define QLEN     512
#define BATCH    4
#define MLEN     512
#define ILEN     1024            // MLEN + QLEN
#define QB       2048            // QLEN*BATCH
#define IBROWS   4096            // ILEN*BATCH
#define FFDIM    4096

typedef long long ll;
typedef unsigned short u16;
typedef __attribute__((ext_vector_type(16))) __bf16 v16bf;
typedef __attribute__((ext_vector_type(8)))  __bf16 v8bf;
typedef __attribute__((ext_vector_type(8)))  float  v8f;

__device__ __forceinline__ u16 f2bf(float f) {
  unsigned u = __float_as_uint(f);
  u += 0x7FFFu + ((u >> 16) & 1u);   // round-to-nearest-even
  return (u16)(u >> 16);
}

// CDNA5 async global->LDS copy (bypasses VGPRs, tracked by ASYNCcnt).
// Generic pointers to LDS carry the LDS byte offset in their low 32 bits
// (ISA 10.2).
__device__ __forceinline__ void async_b128(const u16* g, u16* lds) {
  const unsigned loff = (unsigned)(unsigned long long)lds;
  asm volatile("global_load_async_to_lds_b128 %0, %1, off"
               :: "v"(loff), "v"(g) : "memory");
}
__device__ __forceinline__ void wait_async() {
  asm volatile("s_wait_asynccnt 0x0" ::: "memory");
}

// ---------------- generic batched bf16 WMMA GEMM ----------------
// C[z] = A[z](MxK, row-major bf16) * B[z](NxK row-major, or KxN if bKmajor)^T
// Double-buffered LDS; async global->LDS for the row-major operands.
#define BM 128
#define BN 128
#define BKQ 32
#define BKP 48     // padded LDS K-stride (48*2 = 96 bytes, keeps 32B alignment)

__global__ __launch_bounds__(256) void gemm_bf16_wmma(
    const u16* __restrict__ A, ll lda, ll sA,
    const u16* __restrict__ Bm, ll ldb, ll sB, int bKmajor,
    float* __restrict__ C32, u16* __restrict__ C16, ll ldc, ll sC,
    const float* __restrict__ bias,
    const float* __restrict__ resid, ll ldr,
    int relu, int M, int N, int K)
{
  __shared__ __align__(32) u16 ldsA[2][BM * BKP];
  __shared__ __align__(32) u16 ldsB[2][BN * BKP];

  const int tid  = (int)threadIdx.x;
  const int lane = tid & 31;          // wave32
  const int w    = tid >> 5;          // 8 waves / block
  const int wm   = w >> 2;            // 2 wave rows
  const int wn   = w & 3;             // 4 wave cols
  const int z    = (int)blockIdx.z;
  const int m0   = (int)blockIdx.y * BM;
  const int n0   = (int)blockIdx.x * BN;

  A  += (ll)z * sA;
  Bm += (ll)z * sB;

  v8f acc[4][2];
  #pragma unroll
  for (int i = 0; i < 4; ++i)
    #pragma unroll
    for (int j = 0; j < 2; ++j)
      #pragma unroll
      for (int r = 0; r < 8; ++r) acc[i][j][r] = 0.0f;

  const int kr  = tid >> 3;           // 0..31  (k-major B path)
  const int nc  = (tid & 7) * 16;     // 0..112 (k-major B path)
  const int l15 = lane & 15;
  const int hi  = lane >> 4;

  // Issue one K-tile's loads into buffer nb. A and row-major B use the
  // CDNA5 async-to-LDS path: 2 x 16B chunks per thread per operand.
  auto issue_tile = [&](int kk, int nb) {
    #pragma unroll
    for (int j = 0; j < 2; ++j) {
      const int c   = tid + j * 256;  // chunk 0..511
      const int row = c >> 2;         // 0..127
      const int col = (c & 3) * 8;    // 0,8,16,24 (bf16 elems)
      const int gm  = m0 + row;
      u16* lp = &ldsA[nb][row * BKP + col];
      if (gm < M) async_b128(A + (ll)gm * lda + kk + col, lp);
      else        *(uint4*)lp = make_uint4(0u, 0u, 0u, 0u);
    }
    if (!bKmajor) {
      #pragma unroll
      for (int j = 0; j < 2; ++j) {
        const int c   = tid + j * 256;
        const int row = c >> 2;
        const int col = (c & 3) * 8;
        const int gn  = n0 + row;
        u16* lp = &ldsB[nb][row * BKP + col];
        if (gn < N) async_b128(Bm + (ll)gn * ldb + kk + col, lp);
        else        *(uint4*)lp = make_uint4(0u, 0u, 0u, 0u);
      }
    } else {      // B stored [K,N]: coalesced load along N, transpose into LDS
      uint4 val = make_uint4(0u, 0u, 0u, 0u);
      if (n0 + nc + 15 < N) {
        val = *(const uint4*)(Bm + (ll)(kk + kr) * ldb + n0 + nc);
        if (kk + BKQ < K) __builtin_prefetch(Bm + (ll)(kk + BKQ + kr) * ldb + n0 + nc, 0, 0);
      }
      const u16* pv = (const u16*)&val;
      #pragma unroll
      for (int j = 0; j < 16; ++j) ldsB[nb][(nc + j) * BKP + kr] = pv[j];
    }
  };

  int buf = 0;
  issue_tile(0, buf);
  wait_async();
  __syncthreads();

  for (int kk = 0; kk < K; kk += BKQ) {
    const int nxt = buf ^ 1;
    if (kk + BKQ < K) issue_tile(kk + BKQ, nxt);   // stream next tile while computing

    // Fragment assembly in this code's contiguous K order (shared by A and B, not the ISA table order):
    // A 16x32 bf16: lane holds 16 consecutive K for row = lane%16, K-half = lane/16
    // B 32x16 bf16: lane holds 16 consecutive K for col = lane%16, K-half = lane/16
    v16bf afrag[4], bfrag[2];
    #pragma unroll
    for (int mt = 0; mt < 4; ++mt) {
      const int row = wm * 64 + mt * 16 + l15;
      union { v16bf v; v8bf h[2]; } u;
      u.h[0] = *(const v8bf*)(&ldsA[buf][row * BKP + hi * 16]);
      u.h[1] = *(const v8bf*)(&ldsA[buf][row * BKP + hi * 16 + 8]);
      afrag[mt] = u.v;
    }
    #pragma unroll
    for (int nt = 0; nt < 2; ++nt) {
      const int col = wn * 32 + nt * 16 + l15;
      bfrag[nt] = *(const v16bf*)(&ldsB[buf][col * BKP + hi * 16]);
    }
    #pragma unroll
    for (int mt = 0; mt < 4; ++mt)
      #pragma unroll
      for (int nt = 0; nt < 2; ++nt)
        acc[mt][nt] = __builtin_amdgcn_wmma_f32_16x16x32_bf16(
            false, afrag[mt], false, bfrag[nt], (short)0, acc[mt][nt], false, false);

    wait_async();        // next tile fully landed in LDS (this wave's issues)
    __syncthreads();     // all waves' tiles landed / all reads of old buf done
    buf = nxt;
  }

  // epilogue (C layout: lanes 0-15 N, VGPR r -> M = r + 8*(lane>=16))
  #pragma unroll
  for (int mt = 0; mt < 4; ++mt) {
    #pragma unroll
    for (int nt = 0; nt < 2; ++nt) {
      const int col = n0 + wn * 32 + nt * 16 + l15;
      if (col >= N) continue;
      const int rowbase = m0 + wm * 64 + mt * 16 + hi * 8;
      #pragma unroll
      for (int r = 0; r < 8; ++r) {
        const int row = rowbase + r;
        if (row >= M) continue;
        float v = acc[mt][nt][r];
        if (bias)  v += bias[col];
        if (resid) v += resid[(ll)row * ldr + col];
        if (relu)  v = fmaxf(v, 0.0f);
        const ll o = (ll)z * sC + (ll)row * ldc + col;
        if (C32) C32[o] = v;
        if (C16) C16[o] = f2bf(v);
      }
    }
  }
}

// ---------------- small elementwise / reduction kernels ----------------
__global__ void cvt_kernel(const float* __restrict__ s, u16* __restrict__ d, ll n) {
  for (ll i = (ll)blockIdx.x * blockDim.x + threadIdx.x; i < n; i += (ll)gridDim.x * blockDim.x)
    d[i] = f2bf(s[i]);
}

__global__ void copy16_kernel(const u16* __restrict__ s, u16* __restrict__ d, ll n) {
  for (ll i = (ll)blockIdx.x * blockDim.x + threadIdx.x; i < n; i += (ll)gridDim.x * blockDim.x)
    d[i] = s[i];
}

__global__ void embed_kernel(const int* __restrict__ idxs, const float* __restrict__ emb,
                             float* __restrict__ x32, u16* __restrict__ x16) {
  const ll n = (ll)QB * EMB;
  for (ll i = (ll)blockIdx.x * blockDim.x + threadIdx.x; i < n; i += (ll)gridDim.x * blockDim.x) {
    const int row = (int)(i >> 10);
    const int c   = (int)(i & (EMB - 1));
    const int t   = idxs[row];
    const float v = emb[(ll)t * EMB + c] * 32.0f;   // * sqrt(EMB)
    x32[i] = v; x16[i] = f2bf(v);
  }
}

__global__ void pos_kernel(u16* __restrict__ pos16) {
  const ll n = (ll)ILEN * EMB;
  for (ll i = (ll)blockIdx.x * blockDim.x + threadIdx.x; i < n; i += (ll)gridDim.x * blockDim.x) {
    const int j = (int)(i >> 10);
    const int c = (int)(i & (EMB - 1));
    const float p  = (float)(ILEN - 1 - j);          // descending M+Q-1 .. 0
    const int  cf  = (c < EMB / 2) ? c : (c - EMB / 2);
    const float fr = __powf(10000.0f, -((float)(2 * cf)) / (float)EMB);
    const float v  = (c < EMB / 2) ? __sinf(p * fr) : __cosf(p * fr);
    pos16[i] = f2bf(v);
  }
}

__global__ void qbias_kernel(const float* __restrict__ q32, const float* __restrict__ ub,
                             const float* __restrict__ vb, u16* __restrict__ qu,
                             u16* __restrict__ qv) {
  const ll n = (ll)QB * EMB;
  for (ll i = (ll)blockIdx.x * blockDim.x + threadIdx.x; i < n; i += (ll)gridDim.x * blockDim.x) {
    const int hd = (int)(i & (EMB - 1));
    const float q = q32[i];
    qu[i] = f2bf(q + ub[hd]);
    qv[i] = f2bf(q + vb[hd]);
  }
}

__global__ __launch_bounds__(256) void ln_kernel(const float* __restrict__ src,
    const float* __restrict__ g, const float* __restrict__ b,
    float* __restrict__ o32, u16* __restrict__ o16) {
  __shared__ float red[256];
  const int row = (int)blockIdx.x;
  const int tid = (int)threadIdx.x;
  const float* x = src + (ll)row * EMB;
  float v[4]; float s = 0.0f;
  #pragma unroll
  for (int i = 0; i < 4; ++i) { v[i] = x[tid + i * 256]; s += v[i]; }
  red[tid] = s; __syncthreads();
  for (int o = 128; o > 0; o >>= 1) { if (tid < o) red[tid] += red[tid + o]; __syncthreads(); }
  const float mu = red[0] * (1.0f / EMB); __syncthreads();
  float vs = 0.0f;
  #pragma unroll
  for (int i = 0; i < 4; ++i) { const float d = v[i] - mu; vs += d * d; }
  red[tid] = vs; __syncthreads();
  for (int o = 128; o > 0; o >>= 1) { if (tid < o) red[tid] += red[tid + o]; __syncthreads(); }
  const float rstd = rsqrtf(red[0] * (1.0f / EMB) + 1e-5f);
  #pragma unroll
  for (int i = 0; i < 4; ++i) {
    const int c = tid + i * 256;
    const float y = (v[i] - mu) * rstd * g[c] + b[c];
    o32[(ll)row * EMB + c] = y;
    o16[(ll)row * EMB + c] = f2bf(y);
  }
}

// att[q,i] = softmax_i( (S1[q,i] + S2[q, i+Q-1-q]) * scale ), masked i > q+M
__global__ __launch_bounds__(256) void softmax_kernel(const float* __restrict__ S1,
    const float* __restrict__ S2, u16* __restrict__ att, float scale) {
  __shared__ float red[256];
  const int q   = (int)blockIdx.x;
  const int h   = (int)blockIdx.y;
  const int tid = (int)threadIdx.x;
  const ll  QI  = (ll)QLEN * ILEN;
  const float* s1 = S1 + (ll)h * QI + (ll)q * ILEN;
  const float* s2 = S2 + (ll)h * QI + (ll)q * ILEN;
  u16* a = att + (ll)h * QI + (ll)q * ILEN;
  const int limit = q + MLEN;            // inclusive last valid key
  const int shift = (QLEN - 1) - q;      // rel-shift index offset
  float v[4]; float mx = -3.0e38f;
  #pragma unroll
  for (int i = 0; i < 4; ++i) {
    const int idx = tid + i * 256;
    if (idx <= limit) { v[i] = (s1[idx] + s2[idx + shift]) * scale; mx = fmaxf(mx, v[i]); }
    else v[i] = -3.0e38f;
  }
  red[tid] = mx; __syncthreads();
  for (int o = 128; o > 0; o >>= 1) { if (tid < o) red[tid] = fmaxf(red[tid], red[tid + o]); __syncthreads(); }
  mx = red[0]; __syncthreads();
  float e[4]; float s = 0.0f;
  #pragma unroll
  for (int i = 0; i < 4; ++i) {
    const int idx = tid + i * 256;
    e[i] = (idx <= limit) ? __expf(v[i] - mx) : 0.0f;
    s += e[i];
  }
  red[tid] = s; __syncthreads();
  for (int o = 128; o > 0; o >>= 1) { if (tid < o) red[tid] += red[tid + o]; __syncthreads(); }
  const float inv = 1.0f / red[0];
  #pragma unroll
  for (int i = 0; i < 4; ++i) {
    const int idx = tid + i * 256;
    a[idx] = f2bf(e[i] * inv);
  }
}

// ---------------- host orchestration ----------------
extern "C" void kernel_launch(void* const* d_in, const int* in_sizes, int n_in,
                              void* d_out, int out_size, void* d_ws, size_t ws_size,
                              hipStream_t stream) {
  (void)in_sizes; (void)n_in; (void)out_size; (void)ws_size;
  const int*   idxs      = (const int*)  d_in[0];
  const float* memory    = (const float*)d_in[1];
  const float* embedding = (const float*)d_in[2];
  const float* proj_b    = (const float*)d_in[3];
  const float* u_bias    = (const float*)d_in[4];
  const float* v_bias    = (const float*)d_in[5];
  const float* Wq        = (const float*)d_in[6];
  const float* Wkv       = (const float*)d_in[7];
  const float* Wp        = (const float*)d_in[8];
  const float* fc_w      = (const float*)d_in[9];
  const float* ff1_w     = (const float*)d_in[10];
  const float* ff1_b     = (const float*)d_in[11];
  const float* ff2_w     = (const float*)d_in[12];
  const float* ff2_b     = (const float*)d_in[13];
  const float* ln_g      = (const float*)d_in[14];
  const float* ln_b      = (const float*)d_in[15];

  char* wsp = (char*)d_ws;
  size_t off = 0;
  auto alloc = [&](size_t bytes) -> void* {
    void* p = (void*)(wsp + off);
    off += (bytes + 255) & ~(size_t)255;
    return p;
  };
  u16*   emb16  = (u16*)  alloc((size_t)VOCAB * EMB * 2);
  u16*   Wq16   = (u16*)  alloc((size_t)L_LAYERS * EMB * EMB * 2);
  u16*   Wkv16  = (u16*)  alloc((size_t)L_LAYERS * 2 * EMB * EMB * 2);
  u16*   Wp16   = (u16*)  alloc((size_t)L_LAYERS * EMB * EMB * 2);
  u16*   fcw16  = (u16*)  alloc((size_t)L_LAYERS * EMB * EMB * 2);
  u16*   ff1w16 = (u16*)  alloc((size_t)L_LAYERS * FFDIM * EMB * 2);
  u16*   ff2w16 = (u16*)  alloc((size_t)L_LAYERS * FFDIM * EMB * 2);
  float* x32    = (float*)alloc((size_t)QB * EMB * 4);
  u16*   x16    = (u16*)  alloc((size_t)QB * EMB * 2);
  u16*   pos16  = (u16*)  alloc((size_t)ILEN * EMB * 2);
  u16*   e16    = (u16*)  alloc((size_t)IBROWS * EMB * 2);
  u16*   kv16   = (u16*)  alloc((size_t)IBROWS * 2 * EMB * 2);
  float* q32    = (float*)alloc((size_t)QB * EMB * 4);
  u16*   qu16   = (u16*)  alloc((size_t)QB * EMB * 2);
  u16*   qv16   = (u16*)  alloc((size_t)QB * EMB * 2);
  u16*   p16    = (u16*)  alloc((size_t)ILEN * EMB * 2);
  float* S1     = (float*)alloc((size_t)NH * QLEN * ILEN * 4);
  float* S2     = (float*)alloc((size_t)NH * QLEN * ILEN * 4);
  u16*   att16  = (u16*)  alloc((size_t)NH * QLEN * ILEN * 2);
  float* o32    = (float*)alloc((size_t)QB * EMB * 4);
  u16*   o16    = (u16*)  alloc((size_t)QB * EMB * 2);
  float* t32    = (float*)alloc((size_t)QB * EMB * 4);
  float* h32    = (float*)alloc((size_t)QB * EMB * 4);
  u16*   h16    = (u16*)  alloc((size_t)QB * EMB * 2);
  u16*   f16b   = (u16*)  alloc((size_t)QB * FFDIM * 2);

  auto cvt = [&](const float* s, u16* d, ll n) {
    cvt_kernel<<<dim3(2048), dim3(256), 0, stream>>>(s, d, n);
  };
  auto gemm = [&](const u16* A, ll lda, ll sA, const u16* B, ll ldb, ll sB, int bkm,
                  float* C32, u16* C16, ll ldc, ll sC,
                  const float* bias, const float* resid, ll ldr, int relu,
                  int M, int N, int K, int Z) {
    dim3 g((unsigned)((N + BN - 1) / BN), (unsigned)((M + BM - 1) / BM), (unsigned)Z);
    gemm_bf16_wmma<<<g, dim3(256), 0, stream>>>(A, lda, sA, B, ldb, sB, bkm,
        C32, C16, ldc, sC, bias, resid, ldr, relu, M, N, K);
  };

  // bf16 copies of all weights (recomputed each call; ~12us at HBM rate)
  cvt(embedding, emb16,  (ll)VOCAB * EMB);
  cvt(Wq,    Wq16,   (ll)L_LAYERS * EMB * EMB);
  cvt(Wkv,   Wkv16,  (ll)L_LAYERS * 2 * EMB * EMB);
  cvt(Wp,    Wp16,   (ll)L_LAYERS * EMB * EMB);
  cvt(fc_w,  fcw16,  (ll)L_LAYERS * EMB * EMB);
  cvt(ff1_w, ff1w16, (ll)L_LAYERS * FFDIM * EMB);
  cvt(ff2_w, ff2w16, (ll)L_LAYERS * FFDIM * EMB);

  embed_kernel<<<dim3(2048), dim3(256), 0, stream>>>(idxs, embedding, x32, x16);
  pos_kernel<<<dim3(1024), dim3(256), 0, stream>>>(pos16);

  const float scale = 1.0f / 1073741824.0f;  // 1 / 64^5 (faithful to reference)

  for (int l = 0; l < L_LAYERS; ++l) {
    // e = concat(memory[l], x)  -> bf16 [ILEN*BATCH, EMB]
    cvt(memory + (ll)l * MLEN * BATCH * EMB, e16, (ll)MLEN * BATCH * EMB);
    copy16_kernel<<<dim3(2048), dim3(256), 0, stream>>>(x16, e16 + (ll)MLEN * BATCH * EMB, (ll)QB * EMB);

    // kv = e @ Wkv^T  (bf16 out, layout [i*B+b, 2048])
    gemm(e16, EMB, 0, Wkv16 + (ll)l * 2 * EMB * EMB, EMB, 0, 0,
         nullptr, kv16, 2 * EMB, 0, nullptr, nullptr, 0, 0,
         IBROWS, 2 * EMB, EMB, 1);
    // q = x @ Wq^T (f32)
    gemm(x16, EMB, 0, Wq16 + (ll)l * EMB * EMB, EMB, 0, 0,
         q32, nullptr, EMB, 0, nullptr, nullptr, 0, 0,
         QB, EMB, EMB, 1);
    // p = emb_pos @ Wp^T (bf16)
    gemm(pos16, EMB, 0, Wp16 + (ll)l * EMB * EMB, EMB, 0, 0,
         nullptr, p16, EMB, 0, nullptr, nullptr, 0, 0,
         ILEN, EMB, EMB, 1);
    qbias_kernel<<<dim3(2048), dim3(256), 0, stream>>>(q32, u_bias, v_bias, qu16, qv16);

    for (int b = 0; b < BATCH; ++b) {
      // S1[h][q,i] = (q+u) . k   (batched over h)
      gemm(qu16 + (ll)b * EMB, (ll)BATCH * EMB, HD,
           kv16 + (ll)b * 2 * EMB, (ll)BATCH * 2 * EMB, HD, 0,
           S1, nullptr, ILEN, (ll)QLEN * ILEN, nullptr, nullptr, 0, 0,
           QLEN, ILEN, HD, NH);
      // S2[h][q,j] = (q+v) . p_j (batched over h)
      gemm(qv16 + (ll)b * EMB, (ll)BATCH * EMB, HD,
           p16, EMB, HD, 0,
           S2, nullptr, ILEN, (ll)QLEN * ILEN, nullptr, nullptr, 0, 0,
           QLEN, ILEN, HD, NH);
      softmax_kernel<<<dim3(QLEN, NH), dim3(256), 0, stream>>>(S1, S2, att16, scale);
      // O[h][q,d] = att @ V  (V is [i,d] => K-major B operand)
      gemm(att16, ILEN, (ll)QLEN * ILEN,
           kv16 + (ll)b * 2 * EMB + EMB, (ll)BATCH * 2 * EMB, HD, 1,
           o32 + (ll)b * EMB, o16 + (ll)b * EMB, (ll)BATCH * EMB, HD,
           nullptr, nullptr, 0, 0,
           QLEN, HD, ILEN, NH);
    }

    // fc with residual x, then LN -> h
    gemm(o16, EMB, 0, fcw16 + (ll)l * EMB * EMB, EMB, 0, 0,
         t32, nullptr, EMB, 0, nullptr, x32, EMB, 0,
         QB, EMB, EMB, 1);
    ln_kernel<<<dim3(QB), dim3(256), 0, stream>>>(t32, ln_g + (ll)l * EMB, ln_b + (ll)l * EMB, h32, h16);

    // ff1 (+bias, relu) -> ff2 (+bias, residual h) -> LN -> x
    gemm(h16, EMB, 0, ff1w16 + (ll)l * FFDIM * EMB, EMB, 0, 0,
         nullptr, f16b, FFDIM, 0, ff1_b + (ll)l * FFDIM, nullptr, 0, 1,
         QB, FFDIM, EMB, 1);
    gemm(f16b, FFDIM, 0, ff2w16 + (ll)l * FFDIM * EMB, FFDIM, 0, 0,
         t32, nullptr, EMB, 0, ff2_b + (ll)l * EMB, h32, EMB, 0,
         QB, EMB, FFDIM, 1);
    ln_kernel<<<dim3(QB), dim3(256), 0, stream>>>(t32, ln_g + (ll)l * EMB, ln_b + (ll)l * EMB, x32, x16);
  }

  // tied output projection -> d_out [Q,B,V] f32
  gemm(x16, EMB, 0, emb16, EMB, 0, 0,
       (float*)d_out, nullptr, VOCAB, 0, proj_b, nullptr, 0, 0,
       QB, VOCAB, EMB, 1);
}
